// ClippingAttentionEngine_22436909154791
// MI455X (gfx1250) — matrix operand order repaired and checked by reading, not compiled, not run on hardware
//
#include <hip/hip_runtime.h>

// Problem constants (from reference)
#define Bb 4
#define Ss 1024
#define Dd 1024
#define Hh 16
#define DH 64
#define KSP 32

typedef __attribute__((ext_vector_type(16))) __bf16 bf16x16;
typedef __attribute__((ext_vector_type(8)))  float  f32x8;

// ---------- scalar helpers ----------
__device__ __forceinline__ unsigned short f32_to_bf16(float f) {
  union { float f; unsigned int u; } c; c.f = f;
  unsigned int u = c.u;
  unsigned int r = u + 0x7FFFu + ((u >> 16) & 1u);  // round-to-nearest-even
  return (unsigned short)(r >> 16);
}
__device__ __forceinline__ float bf16_to_f32(unsigned short h) {
  union { unsigned int u; float f; } c; c.u = ((unsigned int)h) << 16; return c.f;
}
__device__ __forceinline__ f32x8 zero8() {
  f32x8 z;
#pragma unroll
  for (int i = 0; i < 8; ++i) z[i] = 0.0f;
  return z;
}

// ---------- WMMA fragment loaders (CDNA5 ISA 7.12.2 layouts, wave32) ----------
// A (16(M) x 32(K) bf16) from (M,K) row-major: two contiguous b128 chunks/lane.
__device__ __forceinline__ bf16x16 load_A_frag(const unsigned short* src, int ld) {
  const int lane = threadIdx.x & 31;
  const int m  = lane & 15;
  const int kb = (lane >> 4) << 3;
  union { uint4 q[2]; bf16x16 v; } u;
  u.q[0] = *(const uint4*)(src + m * ld + kb);
  u.q[1] = *(const uint4*)(src + m * ld + kb + 16);
  return u.v;
}
// B (32(K) x 16(N) bf16) given B-TRANSPOSED storage (N,K) row-major:
// lane[3:0]=N, kb=8*lane[4]; elems 0..7 -> K=kb..kb+7, elems 8..15 -> K=kb+16..kb+23.
__device__ __forceinline__ bf16x16 load_Bt_frag(const unsigned short* src, int ld) {
  const int lane = threadIdx.x & 31;
  const int n  = lane & 15;
  const int kb = (lane >> 4) << 3;
  union { uint4 q[2]; bf16x16 v; } u;
  u.q[0] = *(const uint4*)(src + n * ld + kb);
  u.q[1] = *(const uint4*)(src + n * ld + kb + 16);
  return u.v;
}
__device__ __forceinline__ f32x8 wmma_bf16(bf16x16 a, bf16x16 b, f32x8 c) {
  return __builtin_amdgcn_wmma_f32_16x16x32_bf16(false, a, false, b, (short)0, c, false, false);
}

// ---------- conversion kernel: 8 f32 -> 8 bf16 per thread ----------
__global__ __launch_bounds__(256) void cvt_f32_bf16x8_kernel(const float* __restrict__ in,
                                                             unsigned short* __restrict__ out) {
  int i = blockIdx.x * 256 + threadIdx.x;
  const float4* in4 = (const float4*)in;
  float4 f0 = in4[2 * i], f1 = in4[2 * i + 1];
  uint4 r;
  r.x = (unsigned)f32_to_bf16(f0.x) | ((unsigned)f32_to_bf16(f0.y) << 16);
  r.y = (unsigned)f32_to_bf16(f0.z) | ((unsigned)f32_to_bf16(f0.w) << 16);
  r.z = (unsigned)f32_to_bf16(f1.x) | ((unsigned)f32_to_bf16(f1.y) << 16);
  r.w = (unsigned)f32_to_bf16(f1.z) | ((unsigned)f32_to_bf16(f1.w) << 16);
  ((uint4*)out)[i] = r;
}

// ---------- tiled bf16 GEMM: C(M,N) = A(M,K) * Bt(N,K)^T + bias[n] ----------
// Macro tile 128x64, 8 waves, each wave computes a 2x2 arrangement of 16x16 tiles
// (wm = w&3 -> rows 32*wm..+32, wn = w>>2 -> cols 32*wn..+32): 4 WMMA / 8 b128 loads.
// MODE 0: out bf16, layout (B,H,S,DH)   (Q and K projections)
// MODE 1: out bf16, layout (B,H,DH,S)   (V projection, pre-transposed for P*V)
// MODE 2: out f32,  row-major (M,N)     (final output projection)
template <int MODE>
__global__ __launch_bounds__(256)
void gemm_bf16_kernel(const unsigned short* __restrict__ A,   // (M,K)
                      const unsigned short* __restrict__ Bt,  // (N,K)  == W row-major
                      const float* __restrict__ bias,
                      void* __restrict__ out,
                      int M, int N, int Kd) {
  __shared__ __align__(16) unsigned short As[128 * 32];  // 8 KB
  __shared__ __align__(16) unsigned short Bs[64 * 32];   // 4 KB, (N,K) layout

  const int tid  = threadIdx.x;
  const int lane = tid & 31;
  const int w    = tid >> 5;  // 8 waves
  const int wm   = w & 3;     // row-pair group
  const int wn   = w >> 2;    // col-pair group
  const int n0   = blockIdx.x * 64;
  const int m0   = blockIdx.y * 128;

  f32x8 acc[4];               // [i*2+j]: (Atile i, Btile j)
#pragma unroll
  for (int i = 0; i < 4; ++i) acc[i] = zero8();

  const int arow = tid >> 1, acol = (tid & 1) * 16;  // A tile 128x32: 16 elems/thread
  const int brow = tid >> 2, bcol = (tid & 3) * 8;   // B tile 64x32:   8 elems/thread

  for (int kk = 0; kk < Kd; kk += 32) {
    const uint4* ag = (const uint4*)(A + (size_t)(m0 + arow) * Kd + kk + acol);
    uint4* as = (uint4*)(As + arow * 32 + acol);
    as[0] = ag[0];
    as[1] = ag[1];
    *((uint4*)(Bs + brow * 32 + bcol)) =
        *((const uint4*)(Bt + (size_t)(n0 + brow) * Kd + kk + bcol));
    __syncthreads();

    if (kk + 32 < Kd) {  // prefetch next tiles -> global_prefetch_b8
      __builtin_prefetch(A + (size_t)(m0 + arow) * Kd + kk + 32 + acol, 0, 1);
      __builtin_prefetch(Bt + (size_t)(n0 + brow) * Kd + kk + 32 + bcol, 0, 1);
    }

    bf16x16 a0 = load_A_frag(As + (wm * 32) * 32, 32);
    bf16x16 a1 = load_A_frag(As + (wm * 32 + 16) * 32, 32);
    bf16x16 b0 = load_Bt_frag(Bs + (wn * 32) * 32, 32);
    bf16x16 b1 = load_Bt_frag(Bs + (wn * 32 + 16) * 32, 32);
    acc[0] = wmma_bf16(a0, b0, acc[0]);
    acc[1] = wmma_bf16(a0, b1, acc[1]);
    acc[2] = wmma_bf16(a1, b0, acc[2]);
    acc[3] = wmma_bf16(a1, b1, acc[3]);
    __syncthreads();
  }

#pragma unroll
  for (int i = 0; i < 2; ++i) {
#pragma unroll
    for (int j = 0; j < 2; ++j) {
      f32x8 a = acc[i * 2 + j];
#pragma unroll
      for (int r = 0; r < 8; ++r) {
        int m = m0 + (wm * 2 + i) * 16 + r + ((lane >> 4) << 3);
        int n = n0 + (wn * 2 + j) * 16 + (lane & 15);
        float v = a[r] + bias[n];
        if (MODE == 0) {         // (b, h, s, dh) bf16
          int b = m >> 10, s = m & 1023, h = n >> 6, dh = n & 63;
          ((unsigned short*)out)[((((size_t)b * Hh + h) * Ss) + s) * DH + dh] = f32_to_bf16(v);
        } else if (MODE == 1) {  // (b, h, dh, s) bf16  (transposed V)
          int b = m >> 10, s = m & 1023, h = n >> 6, dh = n & 63;
          ((unsigned short*)out)[(((size_t)b * Hh + h) * DH + dh) * Ss + s] = f32_to_bf16(v);
        } else {                 // row-major f32
          ((float*)out)[(size_t)m * N + n] = v;
        }
      }
    }
  }
}

// ---------- attention: one block per (b, h, 16-query tile) ----------
// Q,K: (B,H,S,DH) bf16; Vt: (B,H,DH,S) bf16; O: (B,S,D) bf16
// Dynamic LDS (96 KB):
//   [0,   64K)  Sl : f32 16x1024 scores; reused for PV split-K partials
//   [64K, 96K)  Pb : bf16 16x1024 probabilities
//   [64K, 80K)  Mlt: u8 transposed mask tile (aliases Pb; dead before Pb written)
__global__ __launch_bounds__(256)
void attention_kernel(const unsigned short* __restrict__ Q,
                      const unsigned short* __restrict__ K,
                      const unsigned short* __restrict__ Vt,
                      const unsigned char* __restrict__ prior_mask,       // (S,S)
                      const int* __restrict__ prior_indices,              // (S,32)
                      const unsigned char* __restrict__ prior_index_mask, // (S,32)
                      const float* __restrict__ u_prev,                   // (B,1)
                      unsigned short* __restrict__ O) {
  extern __shared__ char smem[];
  float*          Sl  = (float*)smem;
  unsigned short* Pb  = (unsigned short*)(smem + 16 * Ss * sizeof(float));
  unsigned char*  Mlt = (unsigned char*)(smem + 16 * Ss * sizeof(float));

  const int q0   = blockIdx.x * 16;
  const int h    = blockIdx.y;
  const int b    = blockIdx.z;
  const int tid  = threadIdx.x, lane = tid & 31, w = tid >> 5;
  const float lam = 10.0f * __expf(-5.0f * u_prev[b]);

  const size_t bh = (size_t)b * Hh + h;
  const unsigned short* Qb  = Q  + bh * Ss * DH;
  const unsigned short* Kb  = K  + bh * Ss * DH;
  const unsigned short* Vtb = Vt + bh * DH * Ss;

  if (lam >= 1.0f) {
    // ---- sparse path: 32 gathered keys per query, one wave per row ----
#pragma unroll
    for (int rr = 0; rr < 2; ++rr) {
      int s = q0 + w * 2 + rr;
      int j = lane;  // key slot
      int idx   = prior_indices[s * KSP + j];
      int valid = prior_index_mask[s * KSP + j];
      const uint4* q4 = (const uint4*)(Qb + (size_t)s * DH);
      const uint4* k4 = (const uint4*)(Kb + (size_t)idx * DH);
      float sc = 0.0f;
#pragma unroll
      for (int c = 0; c < 8; ++c) {
        union { uint4 u; unsigned short hh[8]; } qa, ka;
        qa.u = q4[c]; ka.u = k4[c];
#pragma unroll
        for (int e = 0; e < 8; ++e)
          sc += bf16_to_f32(qa.hh[e]) * bf16_to_f32(ka.hh[e]);
      }
      sc = valid ? sc * 0.125f : -3.0e38f;
      float mx = sc;
#pragma unroll
      for (int off = 16; off; off >>= 1) mx = fmaxf(mx, __shfl_xor(mx, off, 32));
      float p = __expf(sc - mx);
      float sum = p;
#pragma unroll
      for (int off = 16; off; off >>= 1) sum += __shfl_xor(sum, off, 32);
      p /= sum;
      float acc0 = 0.0f, acc1 = 0.0f;
#pragma unroll 4
      for (int jj = 0; jj < 32; ++jj) {
        float pj = __shfl(p, jj, 32);
        int   ij = __shfl(idx, jj, 32);
        acc0 += pj * bf16_to_f32(Vtb[(size_t)(2 * lane)     * Ss + ij]);
        acc1 += pj * bf16_to_f32(Vtb[(size_t)(2 * lane + 1) * Ss + ij]);
      }
      size_t ob = ((size_t)b * Ss + s) * Dd + h * DH + 2 * lane;
      O[ob]     = f32_to_bf16(acc0);
      O[ob + 1] = f32_to_bf16(acc1);
    }
    return;
  }

  // ---- dense path ----
  // Phase 0: stage mask tile TRANSPOSED into LDS: Mlt[key*16 + q] = mask[q0+q][key].
  {
    int q  = tid >> 6;          // 0..3
    int k0 = (tid & 63) * 16;   // 0..1008
#pragma unroll
    for (int i = 0; i < 4; ++i) {
      int qq = q + i * 4;       // rows 0..15
      union { uint4 u; unsigned char c[16]; } mm;
      mm.u = *(const uint4*)(prior_mask + (size_t)(q0 + qq) * Ss + k0);
#pragma unroll
      for (int e = 0; e < 16; ++e) Mlt[(size_t)(k0 + e) * 16 + qq] = mm.c[e];
    }
  }

  // Q A-fragments (16 queries x 64 dh -> two K=32 halves), loaded once.
  bf16x16 a0 = load_A_frag(Qb + (size_t)q0 * DH, DH);
  bf16x16 a1 = load_A_frag(Qb + (size_t)q0 * DH + 32, DH);
  __syncthreads();  // mask tile visible

  // Phase 1: scores. Wave w owns keys [128w,128w+128); K is the (N,K) operand.
  const int key_base = w * 128;
  const int hi8 = (lane >> 4) << 3;
#pragma unroll
  for (int ktile = 0; ktile < 8; ++ktile) {
    int key0 = key_base + ktile * 16;
    bf16x16 b0 = load_Bt_frag(Kb + (size_t)key0 * DH, DH);       // dh 0..31
    bf16x16 b1 = load_Bt_frag(Kb + (size_t)key0 * DH + 32, DH);  // dh 32..63
    f32x8 acc = zero8();
    acc = wmma_bf16(a0, b0, acc);
    acc = wmma_bf16(a1, b1, acc);
    int key = key0 + (lane & 15);
    unsigned long long mk = *(const unsigned long long*)(Mlt + key * 16 + hi8);
#pragma unroll
    for (int r = 0; r < 8; ++r) {
      float bias = ((mk >> (8 * r)) & 0xFFull) ? 0.0f : -lam;
      Sl[(r + hi8) * Ss + key] = acc[r] * 0.125f + bias;
    }
  }
  __syncthreads();

  // Phase 2: softmax. Wave w handles rows 2w, 2w+1; writes bf16 probs to Pb.
#pragma unroll
  for (int rr = 0; rr < 2; ++rr) {
    int row = w * 2 + rr;
    float mx = -3.0e38f;
    for (int i = lane; i < Ss; i += 32) mx = fmaxf(mx, Sl[row * Ss + i]);
#pragma unroll
    for (int off = 16; off; off >>= 1) mx = fmaxf(mx, __shfl_xor(mx, off, 32));
    float sum = 0.0f;
    for (int i = lane; i < Ss; i += 32) sum += __expf(Sl[row * Ss + i] - mx);
#pragma unroll
    for (int off = 16; off; off >>= 1) sum += __shfl_xor(sum, off, 32);
    float inv = 1.0f / sum;
    for (int i = lane; i < Ss; i += 32)
      Pb[row * Ss + i] = f32_to_bf16(__expf(Sl[row * Ss + i] - mx) * inv);
  }
  __syncthreads();  // Pb ready; Sl now dead -> reuse for split-K partials

  // Phase 3: P x V with split-K across all 8 waves.
  // Wave w: column tile ct = w&3 (dh cols 16ct..16ct+16), key half kh = w>>2.
  {
    const int ct = w & 3, kh = w >> 2;
    f32x8 acc = zero8();
#pragma unroll 4
    for (int kt = 0; kt < 16; ++kt) {
      int kti = kh * 16 + kt;
      bf16x16 pf = load_A_frag(Pb + kti * 32, Ss);
      bf16x16 vf = load_Bt_frag(Vtb + (size_t)(ct * 16) * Ss + kti * 32, Ss);
      acc = wmma_bf16(pf, vf, acc);
    }
    if (w >= 4) {  // upper key half: deposit partials into (dead) Sl
#pragma unroll
      for (int r = 0; r < 8; ++r)
        Sl[ct * 256 + (r + hi8) * 16 + (lane & 15)] = acc[r];
    }
    __syncthreads();
    if (w < 4) {   // lower key half: reduce + store
#pragma unroll
      for (int r = 0; r < 8; ++r) {
        int m  = r + hi8;
        int dh = ct * 16 + (lane & 15);
        float v = acc[r] + Sl[ct * 256 + (r + hi8) * 16 + (lane & 15)];
        O[((size_t)b * Ss + q0 + m) * Dd + h * DH + dh] = f32_to_bf16(v);
      }
    }
  }
}

// ---------- host-side orchestration ----------
extern "C" void kernel_launch(void* const* d_in, const int* in_sizes, int n_in,
                              void* d_out, int out_size, void* d_ws, size_t ws_size,
                              hipStream_t stream) {
  const float*         x            = (const float*)d_in[0];
  const unsigned char* prior_mask   = (const unsigned char*)d_in[1];
  const int*           prior_idx    = (const int*)d_in[2];
  const unsigned char* prior_idx_m  = (const unsigned char*)d_in[3];
  const float*         u_prev       = (const float*)d_in[4];
  const float*         Wq = (const float*)d_in[5];  const float* bq = (const float*)d_in[6];
  const float*         Wk = (const float*)d_in[7];  const float* bk = (const float*)d_in[8];
  const float*         Wv = (const float*)d_in[9];  const float* bv = (const float*)d_in[10];
  const float*         Wo = (const float*)d_in[11]; const float* bo = (const float*)d_in[12];

  const size_t XN = (size_t)Bb * Ss * Dd;  // 4M elems
  const size_t WN = (size_t)Dd * Dd;       // 1M elems

  char* ws = (char*)d_ws;
  unsigned short* xb  = (unsigned short*)ws;  ws += XN * 2;
  unsigned short* Wqb = (unsigned short*)ws;  ws += WN * 2;
  unsigned short* Wkb = (unsigned short*)ws;  ws += WN * 2;
  unsigned short* Wvb = (unsigned short*)ws;  ws += WN * 2;
  unsigned short* Wob = (unsigned short*)ws;  ws += WN * 2;
  unsigned short* Qb  = (unsigned short*)ws;  ws += XN * 2;
  unsigned short* Kb  = (unsigned short*)ws;  ws += XN * 2;
  unsigned short* Vtb = (unsigned short*)ws;  ws += XN * 2;
  unsigned short* Ob  = (unsigned short*)ws;  ws += XN * 2;

  // 1) f32 -> bf16 conversions (weights stay row-major (N,K): that IS the B^T operand)
  cvt_f32_bf16x8_kernel<<<dim3((unsigned)(XN / (8 * 256))), 256, 0, stream>>>(x, xb);
  cvt_f32_bf16x8_kernel<<<dim3((unsigned)(WN / (8 * 256))), 256, 0, stream>>>(Wq, Wqb);
  cvt_f32_bf16x8_kernel<<<dim3((unsigned)(WN / (8 * 256))), 256, 0, stream>>>(Wk, Wkb);
  cvt_f32_bf16x8_kernel<<<dim3((unsigned)(WN / (8 * 256))), 256, 0, stream>>>(Wv, Wvb);
  cvt_f32_bf16x8_kernel<<<dim3((unsigned)(WN / (8 * 256))), 256, 0, stream>>>(Wo, Wob);

  // 2) Q/K/V projections (M=B*S=4096, N=D=1024, K=D=1024)
  dim3 ggrid(Dd / 64, (Bb * Ss) / 128);
  gemm_bf16_kernel<0><<<ggrid, 256, 0, stream>>>(xb, Wqb, bq, Qb,  Bb * Ss, Dd, Dd);
  gemm_bf16_kernel<0><<<ggrid, 256, 0, stream>>>(xb, Wkb, bk, Kb,  Bb * Ss, Dd, Dd);
  gemm_bf16_kernel<1><<<ggrid, 256, 0, stream>>>(xb, Wvb, bv, Vtb, Bb * Ss, Dd, Dd);

  // 3) attention (dense WMMA path or sparse gather path, per-batch select)
  //    dynamic LDS: 64 KB scores/partials + 32 KB bf16 P (mask aliases) = 96 KB
  attention_kernel<<<dim3(Ss / 16, Hh, Bb), 256,
                     16 * Ss * sizeof(float) + 16 * Ss * sizeof(unsigned short), stream>>>(
      Qb, Kb, Vtb, prior_mask, prior_idx, prior_idx_m, u_prev, Ob);

  // 4) output projection -> f32 d_out
  gemm_bf16_kernel<2><<<ggrid, 256, 0, stream>>>(Ob, Wob, bo, d_out, Bb * Ss, Dd, Dd);
}
